// MultiExpertDIR_16432544874950
// MI455X (gfx1250) — matrix operand order repaired and checked by reading, not compiled, not run on hardware
//
#include <hip/hip_runtime.h>
#include <cstdint>

// Problem constants (from reference)
constexpr int B = 65536, D = 512, G = 16, H = 128;
constexpr int BM = 64;            // rows of z per block
constexpr int KT = D / 32;        // 16 K-steps of 32
constexpr int NTH = H / 16;       // 8 N-tiles over H
constexpr int W1_FRAGS = G * KT * NTH;   // 2048 fragments
constexpr int FRAG_DW = 32 * 8;          // 256 dwords per fragment (32 lanes x 32B)

typedef __attribute__((ext_vector_type(16))) __bf16 v16bf;
typedef __attribute__((ext_vector_type(8)))  float  v8f;

union Frag {
    uint32_t u[8];
    uint4    q[2];
    v16bf    v;
};

__device__ __forceinline__ uint32_t f2bf(float x) {
    uint32_t u = __float_as_uint(x);
    return (u + 0x7FFFu + ((u >> 16) & 1u)) >> 16;   // round-to-nearest-even
}
__device__ __forceinline__ uint32_t pack2bf(float a, float b) {
    return f2bf(a) | (f2bf(b) << 16);
}

// ---------------------------------------------------------------------------
// Pack W1 (G,D,H) and Wc (D,G) f32 -> bf16 fragments in WMMA B-matrix layout.
// B 32x16 bf16, contiguous k order used here (same as A staging): lanes 0-15 hold K=0..15 (N=lane), lanes 16-31 hold
// K=16..31 (N=lane-16); dword j packs K=2j (lo16), K=2j+1 (hi16).
// One 256-thread block per fragment: thread t -> lane=t/8, j=t%8.
// ---------------------------------------------------------------------------
__global__ void pack_weights(const float* __restrict__ W1,
                             const float* __restrict__ Wc,
                             uint32_t* __restrict__ W1p,
                             uint32_t* __restrict__ Wcp) {
    const int f    = blockIdx.x;
    const int t    = threadIdx.x;
    const int lane = t >> 3;
    const int j    = t & 7;
    const int kbase = (lane < 16) ? 0 : 16;
    const int n     = lane & 15;
    const int k0    = kbase + 2 * j;

    if (f < W1_FRAGS) {
        const int nt = f % NTH;
        const int kt = (f / NTH) % KT;
        const int g  = f / (NTH * KT);
        const int d0 = kt * 32 + k0;
        const int h  = nt * 16 + n;
        const float x0 = W1[((size_t)g * D + d0)     * H + h];
        const float x1 = W1[((size_t)g * D + d0 + 1) * H + h];
        W1p[(size_t)f * FRAG_DW + lane * 8 + j] = pack2bf(x0, x1);
    } else {
        const int kt = f - W1_FRAGS;          // 0..KT-1
        const int d0 = kt * 32 + k0;
        const float x0 = Wc[(size_t)d0       * G + n];
        const float x1 = Wc[(size_t)(d0 + 1) * G + n];
        Wcp[(size_t)kt * FRAG_DW + lane * 8 + j] = pack2bf(x0, x1);
    }
}

// ---------------------------------------------------------------------------
// Fused: logits = z@Wc+bc ; softmax ; per-expert fc1 -> GELU(erf) -> fc2.
// Block = 64 rows of z, 8 waves (4 row-strips x 2 col-halves of H).
// ---------------------------------------------------------------------------
__global__ void moe_fused(const float* __restrict__ z,
                          const float* __restrict__ bc,
                          const float* __restrict__ b1,
                          const float* __restrict__ W2,
                          const float* __restrict__ b2,
                          const uint32_t* __restrict__ W1p,
                          const uint32_t* __restrict__ Wcp,
                          float* __restrict__ out) {
    // z tile staged as bf16 in A-fragment layout: [rs][kt][lane][8 dwords]
    __shared__ __attribute__((aligned(16))) uint32_t zTile[(BM / 16) * KT * FRAG_DW]; // 64 KB
    __shared__ float yacc[2][BM];                                                    // 512 B

    const int t     = threadIdx.x;
    const int wave  = t >> 5;
    const int lane  = t & 31;
    const int waveM = wave & 3;     // row strip 0..3
    const int waveN = wave >> 2;    // H half 0..1
    const int lo    = lane & 15;
    const int hi    = lane >> 4;
    const int row0  = blockIdx.x * BM;

    // ---- stage z tile (f32 -> bf16, scrambled to A-fragment layout) ----
    // A 16x32 bf16, contiguous k order used here (same as packed B): lanes0-15: K=0..15 ; lanes16-31: K=16..31
    for (int i = 0; i < (BM * D / 4) / 256; ++i) {    // 32 iterations of float4
        const int idx = i * 256 + t;
        const int r   = idx >> 7;       // 0..63
        const int d   = (idx & 127) * 4;
        const float4 zv = *(const float4*)(z + (size_t)(row0 + r) * D + d);
        const int rs  = r >> 4, m = r & 15;
        const int kt  = d >> 5, kin = d & 31;
        const int grp = kin >> 3;                          // 0..3
        const int laneD = m + ((grp & 2) ? 16 : 0);
        const int e     = kin & 15;                        // element base within lane
        const int base  = ((rs * KT + kt) * 32 + laneD) * 8 + (e >> 1);
        zTile[base]     = pack2bf(zv.x, zv.y);
        zTile[base + 1] = pack2bf(zv.z, zv.w);
    }
    __syncthreads();

    const int aBase = waveM * KT;     // row-strip base into zTile

    // ---- logits + softmax (waves with waveN==0; wave-uniform branch) ----
    if (waveN == 0) {
        v8f lacc = {};
        for (int kt = 0; kt < KT; ++kt) {
            Frag a, b;
            const uint4* ap = (const uint4*)&zTile[((aBase + kt) * 32 + lane) * 8];
            a.q[0] = ap[0]; a.q[1] = ap[1];
            const uint4* bp = (const uint4*)&Wcp[(size_t)kt * FRAG_DW + lane * 8];
            b.q[0] = bp[0]; b.q[1] = bp[1];
            lacc = __builtin_amdgcn_wmma_f32_16x16x32_bf16(
                false, a.v, false, b.v, (short)0, lacc, false, false);
        }
        const float bcv = bc[lo];
        #pragma unroll
        for (int j = 0; j < 8; ++j) {
            float lg = lacc[j] + bcv;                 // logit for (row j+8*hi, col lo)
            float mx = lg;
            #pragma unroll
            for (int mk = 1; mk < 16; mk <<= 1) mx = fmaxf(mx, __shfl_xor(mx, mk, 32));
            const float ex = __expf(lg - mx);
            float sm = ex;
            #pragma unroll
            for (int mk = 1; mk < 16; mk <<= 1) sm += __shfl_xor(sm, mk, 32);
            const int r = row0 + waveM * 16 + j + 8 * hi;
            out[(size_t)r * G + lo]                 = lg;        // logits
            out[(size_t)B * G + (size_t)r * G + lo] = ex / sm;   // p_g
        }
    }

    // ---- expert loop ----
    const int colBase = waveN * 64;
    for (int g = 0; g < G; ++g) {
        if (t < 2 * BM) ((float*)yacc)[t] = 0.0f;
        __syncthreads();

        v8f acc[4] = {v8f{}, v8f{}, v8f{}, v8f{}};
        // Unroll x4 so the scheduler can keep next iterations' global/LDS
        // fragment loads in flight under the current WMMAs (partial waits),
        // without the VGPR blow-up of a full x16 unroll.
        #pragma unroll 4
        for (int kt = 0; kt < KT; ++kt) {
            Frag a;
            const uint4* ap = (const uint4*)&zTile[((aBase + kt) * 32 + lane) * 8];
            a.q[0] = ap[0]; a.q[1] = ap[1];
            #pragma unroll
            for (int nt = 0; nt < 4; ++nt) {
                Frag b;
                const size_t f = ((size_t)g * KT + kt) * NTH + (waveN * 4 + nt);
                const uint4* bp = (const uint4*)&W1p[f * FRAG_DW + lane * 8];
                b.q[0] = bp[0]; b.q[1] = bp[1];
                acc[nt] = __builtin_amdgcn_wmma_f32_16x16x32_bf16(
                    false, a.v, false, b.v, (short)0, acc[nt], false, false);
            }
        }

        // epilogue: h = gelu(acc + b1), partial y = sum_h h * W2
        #pragma unroll
        for (int j = 0; j < 8; ++j) {
            float s = 0.0f;
            #pragma unroll
            for (int nt = 0; nt < 4; ++nt) {
                const int col = colBase + nt * 16 + lo;
                const float v  = acc[nt][j] + b1[g * H + col];
                const float ge = 0.5f * v * (1.0f + erff(v * 0.7071067811865476f));
                s += ge * W2[g * H + col];
            }
            #pragma unroll
            for (int mk = 1; mk < 16; mk <<= 1) s += __shfl_xor(s, mk, 32);
            if (lo == 0) yacc[waveN][waveM * 16 + j + 8 * hi] = s;
        }
        __syncthreads();

        if (t < BM) {
            const int r = row0 + t;
            out[(size_t)2 * B * G + (size_t)r * G + g] = yacc[0][t] + yacc[1][t] + b2[g];
        }
        __syncthreads();
    }
}

extern "C" void kernel_launch(void* const* d_in, const int* in_sizes, int n_in,
                              void* d_out, int out_size, void* d_ws, size_t ws_size,
                              hipStream_t stream) {
    const float* z  = (const float*)d_in[0];
    const float* Wc = (const float*)d_in[1];
    const float* bc = (const float*)d_in[2];
    const float* W1 = (const float*)d_in[3];
    const float* b1 = (const float*)d_in[4];
    const float* W2 = (const float*)d_in[5];
    const float* b2 = (const float*)d_in[6];
    float* out = (float*)d_out;

    uint32_t* W1p = (uint32_t*)d_ws;                       // 2 MB
    uint32_t* Wcp = W1p + (size_t)W1_FRAGS * FRAG_DW;      // +16 KB

    pack_weights<<<W1_FRAGS + KT, 256, 0, stream>>>(W1, Wc, W1p, Wcp);
    moe_fused<<<B / BM, 256, 0, stream>>>(z, bc, b1, W2, b2, W1p, Wcp, out);
}
